// multihead_self_attention_32882269618586
// MI455X (gfx1250) — compile-verified, matrix operand order checked by reading, not run on hardware
//
#include <hip/hip_runtime.h>

typedef __attribute__((ext_vector_type(16))) _Float16 v16h;
typedef __attribute__((ext_vector_type(8)))  _Float16 v8h;
typedef __attribute__((ext_vector_type(8)))  float    v8f;
typedef __attribute__((ext_vector_type(4)))  unsigned int u32x4;
typedef __attribute__((ext_vector_type(8)))  int      i32x8;
typedef __attribute__((ext_vector_type(4)))  int      i32x4;

#define D_MODEL 1024
#define NHEAD   16
#define DKDIM   64
#define SEQ     2048
#define BATCH   2
#define NTOK    (BATCH * SEQ)   // 4096 tokens

// ---------------------------------------------------------------------------
// Helpers
// ---------------------------------------------------------------------------

// WMMA f16 16x16x32, f32 accumulate.
__device__ __forceinline__ v8f wmma_f16(v16h a, v16h b, v8f c) {
    return __builtin_amdgcn_wmma_f32_16x16x32_f16(
        false, a, false, b, (short)0, c, false, false);
}

// Build a v16h fragment from two contiguous 8-element runs (f16 source).
__device__ __forceinline__ v16h cat16(v8h lo, v8h hi) {
    return __builtin_shufflevector(lo, hi, 0,1,2,3,4,5,6,7,8,9,10,11,12,13,14,15);
}

// Build a v16h fragment from two contiguous 8-float runs (f32 source, convert).
__device__ __forceinline__ v16h cvt16(v8f lo, v8f hi) {
    v16h r;
    #pragma unroll
    for (int j = 0; j < 8; ++j) { r[j] = (_Float16)lo[j]; r[j + 8] = (_Float16)hi[j]; }
    return r;
}

// Low 32 bits of a generic pointer to a __shared__ object = LDS byte address.
__device__ __forceinline__ unsigned lds_addr_of(const void* p) {
    return (unsigned)(size_t)p;
}

// Issue a TDM 2D tile load: tensor is [tensor_dim1 rows][tensor_dim0 cols],
// row stride = stride0 (elements); tile is [tile1 rows][tile0 cols] starting
// at gptr, landing contiguously at lds. D# per CDNA5 ISA 8.3/8.4.
// dsz_log: 0=1B,1=2B,2=4B,3=8B. 6-argument builtin on this toolchain.
__device__ __forceinline__ void tdm_load_2d(unsigned lds,
                                            const void* gptr,
                                            unsigned tensor_dim0,
                                            unsigned tensor_dim1,
                                            unsigned long long stride0,
                                            unsigned tile0, unsigned tile1,
                                            unsigned dsz_log)
{
    unsigned long long ga = (unsigned long long)(size_t)gptr;
    u32x4 g0;
    g0[0] = 1u;                                            // count=1, user D#
    g0[1] = lds;                                           // lds_addr
    g0[2] = (unsigned)(ga & 0xffffffffull);                // global_addr[31:0]
    g0[3] = (unsigned)((ga >> 32) & 0x1ffffffull)          // global_addr[56:32]
          | (2u << 30);                                    // type=2 (image)
    i32x8 g1;
    g1[0] = (int)(dsz_log << 16);                          // mask=0, data_size
    g1[1] = (int)((tensor_dim0 & 0xffffu) << 16);          // dim0[15:0] @bit48
    g1[2] = (int)((tensor_dim0 >> 16) | ((tensor_dim1 & 0xffffu) << 16));
    g1[3] = (int)((tensor_dim1 >> 16) | (tile0 << 16));    // tile_dim0 @bit112
    g1[4] = (int)(tile1 & 0xffffu);                        // tile_dim1; tile_dim2=0
    g1[5] = (int)(stride0 & 0xffffffffull);                // dim0_stride[31:0]
    g1[6] = (int)((stride0 >> 32) & 0xffffull);            // dim0_stride[47:32]
    g1[7] = 0;                                             // dim1_stride=0 (2D)
    i32x4 z4 = {0, 0, 0, 0};
    i32x8 z8 = {0, 0, 0, 0, 0, 0, 0, 0};
    __builtin_amdgcn_tensor_load_to_lds(g0, g1, z4, z4, z8, 0);
}

// ---------------------------------------------------------------------------
// Kernel 1: fused QKV projection (Y = X * W^T) + RoPE on Q/K.
// grid = (NTOK/64, 24). Block computes a 64x128 output tile; each wave owns a
// 64x16 strip (4 accumulators) so its 2 B fragments are reused across 4 A
// tiles: 8 WMMAs per 10 fragment builds per 64-K chunk. TDM double-buffers
// raw f32 tiles X (64x64) and W (128x64). V written transposed [b,h,dk,l].
// ---------------------------------------------------------------------------
__global__ __launch_bounds__(256)
void qkv_rope_kernel(const float* __restrict__ X,
                     const float* __restrict__ Wq,
                     const float* __restrict__ Wk,
                     const float* __restrict__ Wv,
                     _Float16* __restrict__ Qh,
                     _Float16* __restrict__ Kh,
                     _Float16* __restrict__ Vt)
{
    __shared__ __align__(32) float Xb[2][64][64];    // 32 KB
    __shared__ __align__(32) float Wb[2][128][64];   // 64 KB

    const int tid  = threadIdx.x;
    const int lane = tid & 31;
    const int wid  = tid >> 5;
    const int by   = blockIdx.y;
    const int mat  = by >> 3;                  // 0=Q 1=K 2=V
    const int nbase = (by & 7) * 128;
    const float* W = (mat == 0) ? Wq : (mat == 1) ? Wk : Wv;
    const int tokbase = blockIdx.x * 64;
    const int hi = lane >> 4;
    const int c  = lane & 15;

    if (wid == 0) {
        tdm_load_2d(lds_addr_of(&Xb[0][0][0]), &X[(size_t)tokbase * D_MODEL],
                    D_MODEL, NTOK, D_MODEL, 64, 64, 2);
        tdm_load_2d(lds_addr_of(&Wb[0][0][0]), &W[(size_t)nbase * D_MODEL],
                    D_MODEL, D_MODEL, D_MODEL, 64, 128, 2);
    }

    v8f acc[4] = {};
    int cur = 0;
    for (int kc = 0; kc < D_MODEL; kc += 64) {
        if (wid == 0) {
            if (kc + 64 < D_MODEL) {
                tdm_load_2d(lds_addr_of(&Xb[cur ^ 1][0][0]),
                            &X[(size_t)tokbase * D_MODEL + kc + 64],
                            D_MODEL, NTOK, D_MODEL, 64, 64, 2);
                tdm_load_2d(lds_addr_of(&Wb[cur ^ 1][0][0]),
                            &W[(size_t)nbase * D_MODEL + kc + 64],
                            D_MODEL, D_MODEL, D_MODEL, 64, 128, 2);
                __builtin_amdgcn_s_wait_tensorcnt(2);  // current buffer ready
            } else {
                __builtin_amdgcn_s_wait_tensorcnt(0);
            }
        }
        __syncthreads();

        const float* wr = &Wb[cur][wid * 16 + c][0];   // B row (out feature)
        v16h b0 = cvt16(*(const v8f*)&wr[hi * 8],      *(const v8f*)&wr[16 + hi * 8]);
        v16h b1 = cvt16(*(const v8f*)&wr[32 + hi * 8], *(const v8f*)&wr[48 + hi * 8]);
        #pragma unroll
        for (int t = 0; t < 4; ++t) {                  // 4 A tiles share B frags
            const float* xr = &Xb[cur][t * 16 + c][0];
            v16h a0 = cvt16(*(const v8f*)&xr[hi * 8],      *(const v8f*)&xr[16 + hi * 8]);
            v16h a1 = cvt16(*(const v8f*)&xr[32 + hi * 8], *(const v8f*)&xr[48 + hi * 8]);
            acc[t] = wmma_f16(a0, b0, acc[t]);
            acc[t] = wmma_f16(a1, b1, acc[t]);
        }

        __syncthreads();
        cur ^= 1;
    }

    // C layout: element i -> row i + 8*hi, col = lane&15.
    const int nfeat = nbase + wid * 16 + c;
    const int head  = nfeat >> 6;
    const int hd    = nfeat & 63;

    #pragma unroll
    for (int t = 0; t < 4; ++t) {
        #pragma unroll
        for (int i = 0; i < 8; ++i) {
            int row  = t * 16 + i + 8 * hi;
            int tok  = tokbase + row;
            int bidx = tok >> 11;
            int pos  = tok & (SEQ - 1);
            float val = acc[t][i];
            if (mat < 2) {                      // RoPE on Q and K
                float partner = __shfl_xor(val, 1);
                int fi = hd >> 1;
                float ang = (float)pos * powf(10000.0f, -(float)(2 * fi) / (float)DKDIM);
                float cs = cosf(ang), sn = sinf(ang);
                val = (nfeat & 1) ? (partner * sn + val * cs)
                                  : (val * cs - partner * sn);
            }
            if (mat == 2) {                     // V stored transposed [b,h,dk,l]
                Vt[(((size_t)bidx * NHEAD + head) * DKDIM + hd) * SEQ + pos] = (_Float16)val;
            } else {
                _Float16* dst = (mat == 0) ? Qh : Kh;
                dst[(((size_t)bidx * NHEAD + head) * SEQ + pos) * DKDIM + hd] = (_Float16)val;
            }
        }
    }
}

// ---------------------------------------------------------------------------
// Kernel 2: causal flash attention. grid = B*H*(SEQ/128); 8 waves/block.
// TDM stages K [32key][64dk] and V^T [64dk][32key] tiles; all WMMA fragments
// are contiguous 16-byte LDS/global vector loads. Online softmax via 16-lane
// shfl reductions; P re-laid out through per-wave LDS region.
// ---------------------------------------------------------------------------
__global__ __launch_bounds__(256)
void flash_attn_kernel(const _Float16* __restrict__ Qh,
                       const _Float16* __restrict__ Kh,
                       const _Float16* __restrict__ Vt,
                       _Float16* __restrict__ CTX)
{
    __shared__ __align__(32) _Float16 Kt[32][64];    // 4 KB  [key][dk]
    __shared__ __align__(32) _Float16 Vl[64][32];    // 4 KB  [dk][key]
    __shared__ __align__(32) _Float16 Pf[8][32][16]; // 8 KB  per-wave P frags

    const int tid  = threadIdx.x;
    const int lane = tid & 31;
    const int w    = tid >> 5;
    const int qt   = blockIdx.x & 15;
    const int bh   = blockIdx.x >> 4;          // b*NHEAD + h
    const int qb   = qt * 128;
    const size_t kbase = (size_t)bh * SEQ * DKDIM;   // Kh/Qh layout [bh][l][dk]
    const size_t vbase = (size_t)bh * DKDIM * SEQ;   // Vt layout [bh][dk][l]
    const int hi = lane >> 4;
    const int c  = lane & 15;

    // Q fragments (A layout): lane holds row qb + w*16 + c; contiguous loads.
    const _Float16* qp = &Qh[kbase + (size_t)(qb + w * 16 + c) * DKDIM];
    v16h qa0 = cat16(*(const v8h*)&qp[hi * 8],      *(const v8h*)&qp[16 + hi * 8]);
    v16h qa1 = cat16(*(const v8h*)&qp[32 + hi * 8], *(const v8h*)&qp[48 + hi * 8]);

    v8f o0 = {}, o1 = {}, o2 = {}, o3 = {};
    float mrow[8], lrow[8];
    #pragma unroll
    for (int i = 0; i < 8; ++i) { mrow[i] = -1e30f; lrow[i] = 0.0f; }

    const int kend = qb + 128;
    for (int kcb = 0; kcb < kend; kcb += 32) {
        if (w == 0) {
            tdm_load_2d(lds_addr_of(&Kt[0][0]), &Kh[kbase + (size_t)kcb * DKDIM],
                        DKDIM, SEQ, DKDIM, 64, 32, 1);
            tdm_load_2d(lds_addr_of(&Vl[0][0]), &Vt[vbase + kcb],
                        SEQ, DKDIM, SEQ, 32, 64, 1);
            __builtin_amdgcn_s_wait_tensorcnt(0);
        }
        __syncthreads();

        // Scores: two 16-key tiles; contraction dk = 64 (2 WMMAs each).
        v8f s0 = {}, s1 = {};
        {
            const _Float16* kr0 = &Kt[c][0];        // key = kcb + c
            const _Float16* kr1 = &Kt[16 + c][0];   // key = kcb + 16 + c
            v16h kb00 = cat16(*(const v8h*)&kr0[hi * 8],      *(const v8h*)&kr0[16 + hi * 8]);
            v16h kb01 = cat16(*(const v8h*)&kr0[32 + hi * 8], *(const v8h*)&kr0[48 + hi * 8]);
            v16h kb10 = cat16(*(const v8h*)&kr1[hi * 8],      *(const v8h*)&kr1[16 + hi * 8]);
            v16h kb11 = cat16(*(const v8h*)&kr1[32 + hi * 8], *(const v8h*)&kr1[48 + hi * 8]);
            s0 = wmma_f16(qa0, kb00, s0);
            s0 = wmma_f16(qa1, kb01, s0);
            s1 = wmma_f16(qa0, kb10, s1);
            s1 = wmma_f16(qa1, kb11, s1);
        }

        const float scale = 0.125f;  // 1/sqrt(64)
        float cf[8];
        #pragma unroll
        for (int i = 0; i < 8; ++i) {
            int r = i + 8 * hi;
            int q = qb + w * 16 + r;
            float x0 = (kcb + c      <= q) ? s0[i] * scale : -1e30f;
            float x1 = (kcb + 16 + c <= q) ? s1[i] * scale : -1e30f;
            float mx = fmaxf(x0, x1);
            #pragma unroll
            for (int d = 1; d < 16; d <<= 1) mx = fmaxf(mx, __shfl_xor(mx, d));
            float mnew = fmaxf(mrow[i], mx);
            cf[i] = __expf(mrow[i] - mnew);
            float p0 = __expf(x0 - mnew);
            float p1 = __expf(x1 - mnew);
            float rs = p0 + p1;
            #pragma unroll
            for (int d = 1; d < 16; d <<= 1) rs += __shfl_xor(rs, d);
            lrow[i] = lrow[i] * cf[i] + rs;
            mrow[i] = mnew;
            // Scatter P into per-wave A-fragment layout:
            // (row r, key k) -> lane r + 16*((k>>3)&1), half (k&7) + 8*(k>>4).
            int hh = (c >> 3) & 1;
            Pf[w][r + 16 * hh][c & 7]       = (_Float16)p0;
            Pf[w][r + 16 * hh][(c & 7) + 8] = (_Float16)p1;
        }
        #pragma unroll
        for (int i = 0; i < 8; ++i) {
            o0[i] *= cf[i]; o1[i] *= cf[i]; o2[i] *= cf[i]; o3[i] *= cf[i];
        }
        // O += P x V (same-wave DS ordering makes Pf readback safe).
        v16h pa = *(const v16h*)&Pf[w][lane][0];
        {
            const _Float16* v0 = &Vl[c][0];            // dk = nt*16 + c rows
            const _Float16* v1 = &Vl[16 + c][0];
            const _Float16* v2 = &Vl[32 + c][0];
            const _Float16* v3 = &Vl[48 + c][0];
            o0 = wmma_f16(pa, cat16(*(const v8h*)&v0[hi * 8], *(const v8h*)&v0[16 + hi * 8]), o0);
            o1 = wmma_f16(pa, cat16(*(const v8h*)&v1[hi * 8], *(const v8h*)&v1[16 + hi * 8]), o1);
            o2 = wmma_f16(pa, cat16(*(const v8h*)&v2[hi * 8], *(const v8h*)&v2[16 + hi * 8]), o2);
            o3 = wmma_f16(pa, cat16(*(const v8h*)&v3[hi * 8], *(const v8h*)&v3[16 + hi * 8]), o3);
        }
        __syncthreads();
    }

    // Normalize and store ctx as f16 [b, l, h*64 + dk].
    const int bidx = bh >> 4;
    const int hidx = bh & 15;
    #pragma unroll
    for (int i = 0; i < 8; ++i) {
        int r = i + 8 * hi;
        int q = qb + w * 16 + r;
        size_t tok = (size_t)bidx * SEQ + q;
        float inv = 1.0f / lrow[i];
        CTX[tok * D_MODEL + hidx * DKDIM +  0 + c] = (_Float16)(o0[i] * inv);
        CTX[tok * D_MODEL + hidx * DKDIM + 16 + c] = (_Float16)(o1[i] * inv);
        CTX[tok * D_MODEL + hidx * DKDIM + 32 + c] = (_Float16)(o2[i] * inv);
        CTX[tok * D_MODEL + hidx * DKDIM + 48 + c] = (_Float16)(o3[i] * inv);
    }
}

// ---------------------------------------------------------------------------
// Kernel 3: output projection out = ctx * Wo^T, fp32 result.
// grid = (NTOK/64, 8). Same widened 64x128 block tile as kernel 1;
// CTX tile f16 [64][64], Wo tile f32 [128][64], TDM double-buffered.
// ---------------------------------------------------------------------------
__global__ __launch_bounds__(256)
void out_proj_kernel(const _Float16* __restrict__ CTX,
                     const float* __restrict__ Wo,
                     float* __restrict__ Out)
{
    __shared__ __align__(32) _Float16 Ab[2][64][64];  // 16 KB
    __shared__ __align__(32) float    Wb[2][128][64]; // 64 KB

    const int tid  = threadIdx.x;
    const int lane = tid & 31;
    const int wid  = tid >> 5;
    const int tokbase = blockIdx.x * 64;
    const int nbase   = blockIdx.y * 128;
    const int hi = lane >> 4;
    const int c  = lane & 15;

    if (wid == 0) {
        tdm_load_2d(lds_addr_of(&Ab[0][0][0]), &CTX[(size_t)tokbase * D_MODEL],
                    D_MODEL, NTOK, D_MODEL, 64, 64, 1);
        tdm_load_2d(lds_addr_of(&Wb[0][0][0]), &Wo[(size_t)nbase * D_MODEL],
                    D_MODEL, D_MODEL, D_MODEL, 64, 128, 2);
    }

    v8f acc[4] = {};
    int cur = 0;
    for (int kc = 0; kc < D_MODEL; kc += 64) {
        if (wid == 0) {
            if (kc + 64 < D_MODEL) {
                tdm_load_2d(lds_addr_of(&Ab[cur ^ 1][0][0]),
                            &CTX[(size_t)tokbase * D_MODEL + kc + 64],
                            D_MODEL, NTOK, D_MODEL, 64, 64, 1);
                tdm_load_2d(lds_addr_of(&Wb[cur ^ 1][0][0]),
                            &Wo[(size_t)nbase * D_MODEL + kc + 64],
                            D_MODEL, D_MODEL, D_MODEL, 64, 128, 2);
                __builtin_amdgcn_s_wait_tensorcnt(2);
            } else {
                __builtin_amdgcn_s_wait_tensorcnt(0);
            }
        }
        __syncthreads();

        const float* wr = &Wb[cur][wid * 16 + c][0];
        v16h b0 = cvt16(*(const v8f*)&wr[hi * 8],      *(const v8f*)&wr[16 + hi * 8]);
        v16h b1 = cvt16(*(const v8f*)&wr[32 + hi * 8], *(const v8f*)&wr[48 + hi * 8]);
        #pragma unroll
        for (int t = 0; t < 4; ++t) {
            const _Float16* ar = &Ab[cur][t * 16 + c][0];
            v16h a0 = cat16(*(const v8h*)&ar[hi * 8],      *(const v8h*)&ar[16 + hi * 8]);
            v16h a1 = cat16(*(const v8h*)&ar[32 + hi * 8], *(const v8h*)&ar[48 + hi * 8]);
            acc[t] = wmma_f16(a0, b0, acc[t]);
            acc[t] = wmma_f16(a1, b1, acc[t]);
        }

        __syncthreads();
        cur ^= 1;
    }

    #pragma unroll
    for (int t = 0; t < 4; ++t) {
        #pragma unroll
        for (int i = 0; i < 8; ++i) {
            int row = t * 16 + i + 8 * hi;
            Out[(size_t)(tokbase + row) * D_MODEL + nbase + wid * 16 + c] = acc[t][i];
        }
    }
}

// ---------------------------------------------------------------------------
extern "C" void kernel_launch(void* const* d_in, const int* in_sizes, int n_in,
                              void* d_out, int out_size, void* d_ws, size_t ws_size,
                              hipStream_t stream) {
    const float* X  = (const float*)d_in[0];
    const float* Wq = (const float*)d_in[1];
    const float* Wk = (const float*)d_in[2];
    const float* Wv = (const float*)d_in[3];
    const float* Wo = (const float*)d_in[4];
    float* Out = (float*)d_out;

    const size_t n = (size_t)BATCH * NHEAD * SEQ * DKDIM;   // 4M elems
    _Float16* Qh  = (_Float16*)d_ws;
    _Float16* Kh  = Qh + n;
    _Float16* Vt  = Kh + n;           // transposed [b,h,dk,l]
    _Float16* CTX = Vt + n;           // [NTOK][D_MODEL] f16

    qkv_rope_kernel<<<dim3(NTOK / 64, 24), 256, 0, stream>>>(X, Wq, Wk, Wv, Qh, Kh, Vt);
    flash_attn_kernel<<<dim3(BATCH * NHEAD * (SEQ / 128)), 256, 0, stream>>>(Qh, Kh, Vt, CTX);
    out_proj_kernel<<<dim3(NTOK / 64, 8), 256, 0, stream>>>(CTX, Wo, Out);
}
